// AttentionModel_39848706573055
// MI455X (gfx1250) — matrix operand order repaired and checked by reading, not compiled, not run on hardware
//
#include <hip/hip_runtime.h>
#include <stdint.h>

// ---------------------------------------------------------------------------
// Flash attention + dropout for B=16, S=2048, D=128 (fp32 I/O, f16 WMMA core)
// Round 2: DPP row-max reduction, row-sum via ones-WMMA, double-buffered LDS,
//          1 barrier/iter, cheaper dropout RNG folded into epilogue.
// ---------------------------------------------------------------------------

typedef _Float16 v16h __attribute__((ext_vector_type(16)));
typedef _Float16 v8h  __attribute__((ext_vector_type(8)));
typedef float    v8f  __attribute__((ext_vector_type(8)));

constexpr int BATCH   = 16;
constexpr int SEQ     = 2048;
constexpr int HDIM    = 128;
constexpr int BLOCK_M = 128;   // queries per workgroup (8 waves x 16 rows)
constexpr int WAVE_M  = 16;    // query rows per wave
constexpr int BLOCK_N = 32;    // keys per iteration
constexpr int NWAVES  = 8;
constexpr int NCHUNK  = SEQ / BLOCK_N;

// LDS strides in halves; multiples of 16 halves (32 B) so every v16h LDS read
// is 32-byte aligned (clean ds_load_b128 pairs).
constexpr int KSTRIDE  = 144;  // K tile row stride   (128 data + pad)
constexpr int VTSTRIDE = 48;   // V^T tile row stride (32 data + pad)
constexpr int PSTRIDE  = 48;   // P scratch row stride

constexpr float    DROPOUT_P = 0.1f;
constexpr float    INV_KEEP  = 1.0f / (1.0f - DROPOUT_P);
constexpr uint32_t DROP_T16  = 6554u;  // ceil(0.1 * 65536): keep iff draw >= T

__device__ __forceinline__ uint32_t hash_u32(uint32_t x) {
  x ^= x >> 16; x *= 0x7feb352du;
  x ^= x >> 15; x *= 0x846ca68bu;
  x ^= x >> 16;
  return x;
}

// Max-reduction across the 16 lanes of each wave half, entirely in VALU (DPP).
// quad_perm(1,0,3,2) -> quad_perm(2,3,0,1) -> row_half_mirror -> row_mirror.
__device__ __forceinline__ float dpp_max16(float v) {
  float t;
  t = __int_as_float(__builtin_amdgcn_update_dpp(
        __float_as_int(v), __float_as_int(v), 0xB1, 0xF, 0xF, true));
  v = fmaxf(v, t);
  t = __int_as_float(__builtin_amdgcn_update_dpp(
        __float_as_int(v), __float_as_int(v), 0x4E, 0xF, 0xF, true));
  v = fmaxf(v, t);
  t = __int_as_float(__builtin_amdgcn_update_dpp(
        __float_as_int(v), __float_as_int(v), 0x141, 0xF, 0xF, true));
  v = fmaxf(v, t);
  t = __int_as_float(__builtin_amdgcn_update_dpp(
        __float_as_int(v), __float_as_int(v), 0x140, 0xF, 0xF, true));
  v = fmaxf(v, t);
  return v;
}

__global__ __launch_bounds__(256)
void flash_attn_dropout_f16wmma(const float* __restrict__ Q,
                                const float* __restrict__ K,
                                const float* __restrict__ V,
                                float* __restrict__ O) {
  __shared__ _Float16 kTile[2][BLOCK_N * KSTRIDE];   // [n][d] keys, ping-pong
  __shared__ _Float16 vtTile[2][HDIM * VTSTRIDE];    // [d][k] values^T, ping-pong
  __shared__ _Float16 pScratch[NWAVES * WAVE_M * PSTRIDE];  // per-wave P (pre-dropout)

  const int tid  = threadIdx.x;
  const int wave = tid >> 5;
  const int lane = tid & 31;
  const int lrow = lane & 15;  // row/col index within a 16-lane half
  const int lhi  = lane >> 4;  // which 16-lane half

  const int b      = blockIdx.x / (SEQ / BLOCK_M);
  const int mBlock = blockIdx.x % (SEQ / BLOCK_M);
  const int mBase  = mBlock * BLOCK_M + wave * WAVE_M;

  // ---- Q tile as four f16 A-fragments (16x32 K-chunks over D) ----
  // Contiguous k order chosen here (K B-fragments match): lanes 0-15 row M=lane, elems 0-15 = K0..15;
  //           lanes 16-31 same row, K16..31.
  v16h qA[4];
  {
    const float* qRow = Q + ((size_t)b * SEQ + (size_t)(mBase + lrow)) * HDIM;
    #pragma unroll
    for (int c = 0; c < 4; ++c) {
      const int d0 = 32 * c + 16 * lhi;
      #pragma unroll
      for (int e = 0; e < 8; ++e) {
        qA[c][e]     = (_Float16)qRow[d0 + e];
        qA[c][e + 8] = (_Float16)qRow[d0 + 8 + e];
      }
    }
  }

  // Constant ones B-fragment: row sums of P via one WMMA.
  v16h onesB;
  #pragma unroll
  for (int e = 0; e < 16; ++e) onesB[e] = (_Float16)1.0f;

  v8f acc[8] = {};             // 16x128 fp32 output accumulator (8 D-tiles)
  float mrun[8], lrun[8];
  #pragma unroll
  for (int j = 0; j < 8; ++j) { mrun[j] = -3.0e38f; lrun[j] = 0.0f; }

  _Float16* pw = pScratch + (size_t)wave * WAVE_M * PSTRIDE;

  auto stage = [&](int chunk, int buf) {
    const float* kg = K + ((size_t)b * SEQ + (size_t)chunk * BLOCK_N) * HDIM;
    const float* vg = V + ((size_t)b * SEQ + (size_t)chunk * BLOCK_N) * HDIM;
    #pragma unroll
    for (int i = 0; i < (BLOCK_N * HDIM) / 256; ++i) {  // 16 coalesced rounds
      const int idx  = i * 256 + tid;
      const int r    = idx >> 7;    // key row within chunk
      const int dcol = idx & 127;   // head-dim column
      kTile[buf][r * KSTRIDE + dcol]   = (_Float16)kg[idx];
      vtTile[buf][dcol * VTSTRIDE + r] = (_Float16)vg[idx];
    }
  };

  stage(0, 0);
  __syncthreads();

  #pragma unroll 1
  for (int it = 0; it < NCHUNK; ++it) {
    const int cur = it & 1;
    const int kBase = it * BLOCK_N;

    // ---- Stage next chunk into the other buffer (overlaps with compute) ----
    if (it + 1 < NCHUNK) {
      stage(it + 1, cur ^ 1);
      if (it + 2 < NCHUNK) {  // gfx1250 global_prefetch_b8 for chunk after next
        const char* nk = (const char*)(K + ((size_t)b * SEQ + (size_t)(it + 2) * BLOCK_N) * HDIM);
        const char* nv = (const char*)(V + ((size_t)b * SEQ + (size_t)(it + 2) * BLOCK_N) * HDIM);
        __builtin_prefetch(nk + tid * 64, 0, 0);
        __builtin_prefetch(nv + tid * 64, 0, 0);
      }
    }

    // ---- Scores: S = Q K^T over 32 keys (two 16x16 tiles, 4 K-chunks) ----
    const _Float16* kt = kTile[cur];
    v8f s0 = {};
    v8f s1 = {};
    #pragma unroll
    for (int c = 0; c < 4; ++c) {
      const v16h kb0 = *(const v16h*)(kt + (size_t)lrow * KSTRIDE + 32 * c + 16 * lhi);
      const v16h kb1 = *(const v16h*)(kt + (size_t)(16 + lrow) * KSTRIDE + 32 * c + 16 * lhi);
      s0 = __builtin_amdgcn_wmma_f32_16x16x32_f16(false, qA[c], false, kb0,
                                                  (short)0, s0, false, false);
      s1 = __builtin_amdgcn_wmma_f32_16x16x32_f16(false, qA[c], false, kb1,
                                                  (short)0, s1, false, false);
    }

    // ---- Online softmax (pre-dropout P -> per-wave LDS scratch) ----
    float scArr[8];
    #pragma unroll
    for (int j = 0; j < 8; ++j) {
      const float rmax = dpp_max16(fmaxf(s0[j], s1[j]));
      const float mnew = fmaxf(mrun[j], rmax);
      const float sc   = __expf(mrun[j] - mnew);
      mrun[j]  = mnew;
      scArr[j] = sc;

      const float p0 = __expf(s0[j] - mnew);
      const float p1 = __expf(s1[j] - mnew);
      pw[(j + 8 * lhi) * PSTRIDE + lrow]      = (_Float16)p0;
      pw[(j + 8 * lhi) * PSTRIDE + 16 + lrow] = (_Float16)p1;

      #pragma unroll
      for (int t = 0; t < 8; ++t) acc[t][j] *= sc;  // rescale running output
    }

    // Same-wave LDS is in-order (DScnt in-order per ISA); just pin program order.
    asm volatile("" ::: "memory");

    // ---- Re-read P in A-fragment layout (C->A transpose via LDS) ----
    const v8h plo = *(const v8h*)(pw + (size_t)lrow * PSTRIDE + 8 * lhi);
    const v8h phi = *(const v8h*)(pw + (size_t)lrow * PSTRIDE + 8 * lhi + 16);
    v16h pA = __builtin_shufflevector(plo, phi,
        0, 1, 2, 3, 4, 5, 6, 7, 8, 9, 10, 11, 12, 13, 14, 15);

    // ---- Row sums of pre-dropout P via ones-WMMA: l update ----
    v8f rsum = {};
    rsum = __builtin_amdgcn_wmma_f32_16x16x32_f16(false, pA, false, onesB,
                                                  (short)0, rsum, false, false);
    #pragma unroll
    for (int j = 0; j < 8; ++j) lrun[j] = lrun[j] * scArr[j] + rsum[j];

    // ---- Dropout mask applied to the A-fragment (1/(1-p) folded into epilogue)
    {
      const uint32_t q    = (uint32_t)(mBase + lrow);      // A-layout row
      const uint32_t base = ((uint32_t)b * SEQ + q) * SEQ + (uint32_t)kBase + 8u * lhi;
      #pragma unroll
      for (int e = 0; e < 8; ++e) {
        const uint32_t h = hash_u32(base + e);
        if ((h & 0xFFFFu) < DROP_T16) pA[e]     = (_Float16)0.0f;  // k = 8*lhi+e
        if ((h >> 16)     < DROP_T16) pA[e + 8] = (_Float16)0.0f;  // k = 16+8*lhi+e
      }
    }

    // ---- O += P * V : A = P(16x32), B = V chunk (32 x 16 per D-tile) ----
    const _Float16* vt = vtTile[cur];
    #pragma unroll
    for (int t = 0; t < 8; ++t) {
      const v8h vlo = *(const v8h*)(vt + (size_t)(t * 16 + lrow) * VTSTRIDE + 8 * lhi);
      const v8h vhi = *(const v8h*)(vt + (size_t)(t * 16 + lrow) * VTSTRIDE + 8 * lhi + 16);
      const v16h vb = __builtin_shufflevector(vlo, vhi,
          0, 1, 2, 3, 4, 5, 6, 7, 8, 9, 10, 11, 12, 13, 14, 15);
      acc[t] = __builtin_amdgcn_wmma_f32_16x16x32_f16(false, pA, false, vb,
                                                      (short)0, acc[t], false, false);
    }

    __syncthreads();  // next buffer staged by all; current tile reads done
  }

  // ---- Normalize: 1/l and the dropout 1/(1-p) rescale together ----
  #pragma unroll
  for (int j = 0; j < 8; ++j) {
    const float inv = INV_KEEP / lrun[j];
    const int q = mBase + j + 8 * lhi;
    float* orow = O + ((size_t)b * SEQ + q) * HDIM;
    #pragma unroll
    for (int t = 0; t < 8; ++t) {
      orow[t * 16 + lrow] = acc[t][j] * inv;
    }
  }
}

extern "C" void kernel_launch(void* const* d_in, const int* in_sizes, int n_in,
                              void* d_out, int out_size, void* d_ws, size_t ws_size,
                              hipStream_t stream) {
  (void)in_sizes; (void)n_in; (void)d_ws; (void)ws_size; (void)out_size;
  const float* Q = (const float*)d_in[0];
  const float* K = (const float*)d_in[1];
  const float* V = (const float*)d_in[2];
  float* O = (float*)d_out;

  dim3 grid(BATCH * (SEQ / BLOCK_M));  // 256 workgroups
  dim3 block(256);                      // 8 wave32 waves
  hipLaunchKernelGGL(flash_attn_dropout_f16wmma, grid, block, 0, stream, Q, K, V, O);
}
